// TripleDecisionGraph_78546361909960
// MI455X (gfx1250) — matrix operand order repaired and checked by reading, not compiled, not run on hardware
//
#include <hip/hip_runtime.h>
#include <hip/hip_bf16.h>
#include <stdint.h>

#define NN 8192
#define DD 128

typedef _Float16 f16;
typedef __attribute__((ext_vector_type(16))) _Float16 v16h;
typedef __attribute__((ext_vector_type(8)))  _Float16 h8;
typedef __attribute__((ext_vector_type(8)))  float v8f;
typedef __attribute__((ext_vector_type(4)))  unsigned int u32x4;
typedef __attribute__((ext_vector_type(8)))  int i32x8;
typedef __attribute__((ext_vector_type(4)))  int i32x4;

static __device__ __forceinline__ v8f wmma_f16f32(v16h a, v16h b, v8f c) {
  return __builtin_amdgcn_wmma_f32_16x16x32_f16(false, a, false, b, (short)0, c, false, false);
}

static __device__ __forceinline__ v8f zero_v8f() {
  v8f z = {0.f, 0.f, 0.f, 0.f, 0.f, 0.f, 0.f, 0.f};
  return z;
}

static __device__ __forceinline__ v16h cat16(h8 lo, h8 hi) {
  return __builtin_shufflevector(lo, hi, 0, 1, 2, 3, 4, 5, 6, 7, 8, 9, 10, 11, 12, 13, 14, 15);
}

// A fragment (16x32 f16) per ISA 7.12.2: lane m = lane&15; K halves:
// e=0..7 -> K = kbase + (lane>>4)*8 + e ; e=8..15 -> K = kbase+16 + (lane>>4)*8 + (e-8)
static __device__ __forceinline__ v16h load_a_rm(const f16* src, int ld, int rbase, int kbase, int lane) {
  const f16* p = src + (size_t)(rbase + (lane & 15)) * ld + kbase + ((lane >> 4) << 3);
  h8 lo = *(const h8*)p;
  h8 hi = *(const h8*)(p + 16);
  return cat16(lo, hi);
}

// B fragment (32x16): lane n = lane&15; e=0..7 -> K = (lane>>4)*8 + e ; e=8..15 -> K = 16 + (lane>>4)*8 + (e-8)
// value B[k][n] = src[(nbase+n)*ld + kbase + k]
static __device__ __forceinline__ v16h load_b_nm(const f16* src, int ld, int nbase, int kbase, int lane) {
  const f16* p = src + (size_t)(nbase + (lane & 15)) * ld + kbase + ((lane >> 4) << 3);
  h8 lo = *(const h8*)p;
  h8 hi = *(const h8*)(p + 16);
  return cat16(lo, hi);
}

// ---- Tensor Data Mover: 2-D tile load into LDS (D# per ISA ch.8) ----
static __device__ __forceinline__ void tdm_load_2d(uint32_t lds_off, const void* gptr,
                                                   uint32_t tile_d0, uint32_t tile_d1,
                                                   uint32_t tensor_d0, uint32_t tensor_d1,
                                                   uint32_t stride0_elems) {
  uint64_t ga = (uint64_t)(uintptr_t)gptr;
  u32x4 g0;
  g0.x = 1u;                                                // count=1, user mode
  g0.y = lds_off;                                           // lds_addr (bytes)
  g0.z = (uint32_t)ga;                                      // global_addr[31:0]
  g0.w = (uint32_t)((ga >> 32) & 0x01FFFFFFu) | (2u << 30); // addr[56:32], type=2
  i32x8 g1;
  g1[0] = (int)(1u << 16);                                  // data_size=1 (2 bytes)
  g1[1] = (int)((tensor_d0 & 0xFFFFu) << 16);               // tensor_dim0[15:0]
  g1[2] = (int)(((tensor_d0 >> 16) & 0xFFFFu) | ((tensor_d1 & 0xFFFFu) << 16));
  g1[3] = (int)(((tensor_d1 >> 16) & 0xFFFFu) | ((tile_d0 & 0xFFFFu) << 16));
  g1[4] = (int)(tile_d1 & 0xFFFFu);                         // tile_dim1, tile_dim2=0
  g1[5] = (int)stride0_elems;                               // tensor_dim0_stride lo32
  g1[6] = 0;
  g1[7] = 0;
  i32x4 z4 = {0, 0, 0, 0};
#if __has_include(<hip/amd_detail/amd_gfx1250_TDM.h>)
  i32x8 z8 = {0, 0, 0, 0, 0, 0, 0, 0};
  __builtin_amdgcn_tensor_load_to_lds(g0, g1, z4, z4, z8, 0);
#else
  __builtin_amdgcn_tensor_load_to_lds(g0, g1, z4, z4, 0);
#endif
}

// =====================================================================
// Kernel A: per-row QKV projection (f32->f16), f16 copies of E (normal +
// transposed), transposed V, and 1/max(||E_r||,1e-8).
// =====================================================================
__global__ __launch_bounds__(128) void qkv_norm_kernel(
    const float* __restrict__ E,
    const float* __restrict__ wq, const float* __restrict__ bq,
    const float* __restrict__ wk, const float* __restrict__ bk,
    const float* __restrict__ wv, const float* __restrict__ bv,
    f16* __restrict__ Eh, f16* __restrict__ EhT,
    f16* __restrict__ Qh, f16* __restrict__ Kh,
    f16* __restrict__ VhT, float* __restrict__ invn) {
  __shared__ float sE[DD];
  __shared__ float red[DD];
  const int r = blockIdx.x;
  const int d = threadIdx.x;
  const float e = E[(size_t)r * DD + d];
  sE[d] = e;
  red[d] = e * e;
  __syncthreads();
  for (int o = 64; o > 0; o >>= 1) {
    if (d < o) red[d] += red[d + o];
    __syncthreads();
  }
  float q = bq[d], k = bk[d], v = bv[d];
#pragma unroll 8
  for (int i = 0; i < DD; ++i) {
    const float ei = sE[i];
    q += ei * wq[i * DD + d];
    k += ei * wk[i * DD + d];
    v += ei * wv[i * DD + d];
  }
  Eh[(size_t)r * DD + d] = (f16)e;
  EhT[(size_t)d * NN + r] = (f16)e;
  Qh[(size_t)r * DD + d] = (f16)q;
  Kh[(size_t)r * DD + d] = (f16)k;
  VhT[(size_t)d * NN + r] = (f16)v;
  if (d == 0) invn[r] = 1.0f / fmaxf(sqrtf(red[0]), 1e-8f);
}

// issue this wave's 4 column tiles (E, K, E^T, V^T) for column block cb
static __device__ __forceinline__ void issue_tiles4(
    f16* dst4, int cb, const f16* Eh, const f16* Kh, const f16* EhT, const f16* VhT) {
  tdm_load_2d((uint32_t)(uintptr_t)(dst4 + 0 * 32 * DD), Eh + (size_t)cb * DD, DD, 32, DD, NN, DD);
  tdm_load_2d((uint32_t)(uintptr_t)(dst4 + 1 * 32 * DD), Kh + (size_t)cb * DD, DD, 32, DD, NN, DD);
  tdm_load_2d((uint32_t)(uintptr_t)(dst4 + 2 * 32 * DD), EhT + cb, 32, DD, NN, DD, NN);
  tdm_load_2d((uint32_t)(uintptr_t)(dst4 + 3 * 32 * DD), VhT + cb, 32, DD, NN, DD, NN);
}

// =====================================================================
// Kernel B: fused flash-style pass. One 16-row tile per BLOCK; the two
// waves split the 256 column blocks (even/odd), each with wave-private
// double-buffered TDM staging -> zero barriers in the main loop. Partials
// (pos/neg sums, counts, online-softmax m/l/acc) merge in-block via LDS.
// =====================================================================
__global__ __launch_bounds__(64) void fused_aggregate_kernel(
    const float* __restrict__ adj,
    const f16* __restrict__ Eh, const f16* __restrict__ EhT,
    const f16* __restrict__ Qh, const f16* __restrict__ Kh,
    const f16* __restrict__ VhT,
    const float* __restrict__ invn,
    float* __restrict__ posO, float* __restrict__ bndO,
    float* __restrict__ negO, float* __restrict__ anyN) {
  // [wave][buf][tile: 0=E,1=K,2=Et,3=Vt] -> 128 KB
  __shared__ __align__(32) f16 sT[2][2][4][32 * DD];
  __shared__ __align__(32) f16 sMask[2][3][16 * 32];

  const int lane = threadIdx.x & 31;
  const int wave = threadIdx.x >> 5;
  const int kh = lane >> 4;
  const int ln = lane & 15;
  const int rowbase = blockIdx.x * 16;  // both waves share this row tile

  float rinv[8];
#pragma unroll
  for (int i = 0; i < 8; ++i) rinv[i] = invn[rowbase + kh * 8 + i];

  v8f accP[8], accN[8], accB[8];
#pragma unroll
  for (int i = 0; i < 8; ++i) { accP[i] = zero_v8f(); accN[i] = zero_v8f(); accB[i] = zero_v8f(); }
  float mrow[8], lrow[8], cP[8], cN[8], cA[8];
#pragma unroll
  for (int i = 0; i < 8; ++i) { mrow[i] = -3.0e38f; lrow[i] = 0.f; cP[i] = 0.f; cN[i] = 0.f; cA[i] = 0.f; }

  const float qscale = 0.0883883476483184f;  // 1/sqrt(128)
  f16* myP = sMask[wave][0];
  f16* myN = sMask[wave][1];
  f16* myW = sMask[wave][2];

  // prologue: stage this wave's first column block into buffer 0
  issue_tiles4(&sT[wave][0][0][0], wave * 32, Eh, Kh, EhT, VhT);

  for (int it = 0; it < 128; ++it) {
    const int buf = it & 1;
    const int cb = (2 * it + wave) * 32;  // this wave's column block
    const bool more = (it + 1) < 128;
    // all ds reads of buf^1 retired before the TDM may overwrite it
    asm volatile("s_wait_dscnt 0x0" ::: "memory");
    if (more) {
      issue_tiles4(&sT[wave][buf ^ 1][0][0], cb + 64, Eh, Kh, EhT, VhT);
      __builtin_amdgcn_s_wait_tensorcnt(4);  // current quad done; next in flight
    } else {
      __builtin_amdgcn_s_wait_tensorcnt(0);
    }
    const f16* bE = sT[wave][buf][0];
    const f16* bK = sT[wave][buf][1];
    const f16* bEt = sT[wave][buf][2];
    const f16* bVt = sT[wave][buf][3];

    // prefetch this wave's next adj tile (one lane per row -> one 128B line)
    if (ln == 0 && cb + 64 < NN) {
#pragma unroll
      for (int i = 0; i < 8; ++i)
        __builtin_prefetch(&adj[(size_t)(rowbase + kh * 8 + i) * NN + cb + 64], 0, 1);
    }

    // 16x32 score tiles: sim (E.E^T) and s (Q.K^T), K-reduced over D=128.
    // Row A-fragments reload from L2-resident Eh/Qh to keep VGPRs lower.
    v8f sc0 = zero_v8f(), sc1 = zero_v8f(), sq0 = zero_v8f(), sq1 = zero_v8f();
#pragma unroll
    for (int s = 0; s < 4; ++s) {
      const v16h aEs = load_a_rm(Eh, DD, rowbase, s * 32, lane);
      const v16h aQs = load_a_rm(Qh, DD, rowbase, s * 32, lane);
      sc0 = wmma_f16f32(aEs, load_b_nm(bE, DD, 0, s * 32, lane), sc0);
      sc1 = wmma_f16f32(aEs, load_b_nm(bE, DD, 16, s * 32, lane), sc1);
      sq0 = wmma_f16f32(aQs, load_b_nm(bK, DD, 0, s * 32, lane), sq0);
      sq1 = wmma_f16f32(aQs, load_b_nm(bK, DD, 16, s * 32, lane), sq1);
    }

    float sv0[8], sv1[8];
#pragma unroll
    for (int st = 0; st < 2; ++st) {
      const v8f* scp = st ? &sc1 : &sc0;
      const v8f* sqp = st ? &sq1 : &sq0;
      const int col = cb + st * 16 + ln;
      const float cinv = invn[col];
      float* svp = st ? sv1 : sv0;
#pragma unroll
      for (int i = 0; i < 8; ++i) {
        const int row = rowbase + kh * 8 + i;
        const float a = adj[(size_t)row * NN + col];
        const bool nb = (a != 0.0f);
        const float sim = (*scp)[i] * (rinv[i] * cinv);
        const float sv = (*sqp)[i] * qscale;
        const bool isP = nb && (sim >= 0.7f);
        const bool isNg = nb && (sim <= 0.3f);
        const bool isB = nb && (sim > 0.3f) && (sim < 0.7f);
        cA[i] += nb ? 1.f : 0.f;
        cP[i] += isP ? 1.f : 0.f;
        cN[i] += isNg ? 1.f : 0.f;
        const int mi = (kh * 8 + i) * 32 + st * 16 + ln;
        myP[mi] = isP ? (f16)1.0f : (f16)0.0f;
        myN[mi] = isNg ? (f16)1.0f : (f16)0.0f;
        svp[i] = isB ? sv : -3.0e38f;
      }
    }

    // online softmax (row max across 16-lane groups); rescale only on change
#pragma unroll
    for (int i = 0; i < 8; ++i) {
      float tm = fmaxf(sv0[i], sv1[i]);
#pragma unroll
      for (int o = 1; o < 16; o <<= 1) tm = fmaxf(tm, __shfl_xor(tm, o, 32));
      const float oldm = mrow[i];
      float e0 = 0.f, e1 = 0.f;
      if (tm > -1.0e37f) {
        if (tm > oldm) {
          const float scale = __expf(oldm - tm);  // ->0 if oldm was sentinel
          lrow[i] *= scale;
#pragma unroll
          for (int nt = 0; nt < 8; ++nt) accB[nt][i] *= scale;
          mrow[i] = tm;
        }
        const float newm = mrow[i];
        e0 = (sv0[i] > -1.0e37f) ? __expf(sv0[i] - newm) : 0.f;
        e1 = (sv1[i] > -1.0e37f) ? __expf(sv1[i] - newm) : 0.f;
      }
      lrow[i] += e0 + e1;
      const int mi = (kh * 8 + i) * 32 + ln;
      myW[mi] = (f16)e0;
      myW[mi + 16] = (f16)e1;
    }

    // fold this 32-column block into the region accumulators (K=32 WMMA)
    const v16h aP = load_a_rm(myP, 32, 0, 0, lane);
    const v16h aNg = load_a_rm(myN, 32, 0, 0, lane);
    const v16h aW = load_a_rm(myW, 32, 0, 0, lane);
#pragma unroll
    for (int nt = 0; nt < 8; ++nt) {
      const v16h bEf = load_b_nm(bEt, 32, nt * 16, 0, lane);
      accP[nt] = wmma_f16f32(aP, bEf, accP[nt]);
      accN[nt] = wmma_f16f32(aNg, bEf, accN[nt]);
      const v16h bVf = load_b_nm(bVt, 32, nt * 16, 0, lane);
      accB[nt] = wmma_f16f32(aW, bVf, accB[nt]);
    }
  }

  // ---- in-block merge of the two column-split partials (via LDS) ----
  float* mb = (float*)&sT[1][0][0][0];  // 3 * 2048 f32 = 24 KB (tiles are dead)
  float* ms = (float*)&sT[1][1][0][0];  // 5 * 8 * 32 f32 stats
  if (wave == 1) {
#pragma unroll
    for (int nt = 0; nt < 8; ++nt)
#pragma unroll
      for (int i = 0; i < 8; ++i) {
        const int x = (nt * 8 + i) * 32 + lane;
        mb[0 * 2048 + x] = accP[nt][i];
        mb[1 * 2048 + x] = accN[nt][i];
        mb[2 * 2048 + x] = accB[nt][i];
      }
#pragma unroll
    for (int i = 0; i < 8; ++i) {
      ms[(0 * 8 + i) * 32 + lane] = mrow[i];
      ms[(1 * 8 + i) * 32 + lane] = lrow[i];
      ms[(2 * 8 + i) * 32 + lane] = cP[i];
      ms[(3 * 8 + i) * 32 + lane] = cN[i];
      ms[(4 * 8 + i) * 32 + lane] = cA[i];
    }
  }
  __syncthreads();
  if (wave == 0) {
#pragma unroll
    for (int i = 0; i < 8; ++i) {
      const float m1 = ms[(0 * 8 + i) * 32 + lane];
      const float l1 = ms[(1 * 8 + i) * 32 + lane];
      cP[i] += ms[(2 * 8 + i) * 32 + lane];
      cN[i] += ms[(3 * 8 + i) * 32 + lane];
      cA[i] += ms[(4 * 8 + i) * 32 + lane];
      const float m0 = mrow[i];
      const float m = fmaxf(m0, m1);
      float s0 = 0.f, s1 = 0.f;
      if (m > -1.0e37f) {
        s0 = (m0 > -1.0e37f) ? __expf(m0 - m) : 0.f;
        s1 = (m1 > -1.0e37f) ? __expf(m1 - m) : 0.f;
      }
      lrow[i] = lrow[i] * s0 + l1 * s1;
#pragma unroll
      for (int nt = 0; nt < 8; ++nt) {
        const int x = (nt * 8 + i) * 32 + lane;
        accP[nt][i] += mb[0 * 2048 + x];
        accN[nt][i] += mb[1 * 2048 + x];
        accB[nt][i] = accB[nt][i] * s0 + mb[2 * 2048 + x] * s1;
      }
    }

    // finalize: reduce per-row counts / denominators across lanes
    float dP[8], dNg[8], dB[8], anyf[8];
#pragma unroll
    for (int i = 0; i < 8; ++i) {
      float a = cP[i], b = cN[i], c = cA[i], d = lrow[i];
#pragma unroll
      for (int o = 1; o < 16; o <<= 1) {
        a += __shfl_xor(a, o, 32);
        b += __shfl_xor(b, o, 32);
        c += __shfl_xor(c, o, 32);
        d += __shfl_xor(d, o, 32);
      }
      dP[i] = 1.0f / fmaxf(a, 1.0f);
      dNg[i] = 0.1f / fmaxf(b, 1.0f);  // LAM folded in
      dB[i] = 1.0f / fmaxf(d, 1.0e-30f);
      anyf[i] = (c > 0.f) ? 1.f : 0.f;
    }
#pragma unroll
    for (int nt = 0; nt < 8; ++nt) {
#pragma unroll
      for (int i = 0; i < 8; ++i) {
        const size_t idx = (size_t)(rowbase + kh * 8 + i) * DD + nt * 16 + ln;
        posO[idx] = accP[nt][i] * dP[i];
        negO[idx] = accN[nt][i] * dNg[i];
        bndO[idx] = accB[nt][i] * dB[i];
      }
    }
    if (ln == 0) {
#pragma unroll
      for (int i = 0; i < 8; ++i) anyN[rowbase + kh * 8 + i] = anyf[i];
    }
  }
}

// =====================================================================
// Kernel C: gated fusion MLP per 16-row tile, WMMA for concat[16x512]@g1.
// =====================================================================
__global__ __launch_bounds__(128) void gate_fuse_kernel(
    const float* __restrict__ E, const float* __restrict__ posW,
    const float* __restrict__ bndW, const float* __restrict__ negW,
    const float* __restrict__ anyN, const f16* __restrict__ g1T,
    const float* __restrict__ g1b, const float* __restrict__ g2w,
    const float* __restrict__ g2b, float* __restrict__ agg) {
  __shared__ __align__(32) f16 sC[16 * 512];
  __shared__ float sH[16 * 256];
  __shared__ float sG[16 * 4];
  const int tid = threadIdx.x;
  const int lane = tid & 31, wave = tid >> 5;
  const int kh = lane >> 4, ln = lane & 15;
  const int rowbase = blockIdx.x * 16;

  for (int t = tid; t < 16 * 512; t += 128) {
    const int r = t >> 9, c = t & 511;
    const float* src = (c < 128) ? E : (c < 256) ? posW : (c < 384) ? bndW : negW;
    sC[t] = (f16)src[(size_t)(rowbase + r) * DD + (c & 127)];
  }
  __syncthreads();

  v8f acc[4];
#pragma unroll
  for (int j = 0; j < 4; ++j) acc[j] = zero_v8f();
#pragma unroll
  for (int ks = 0; ks < 16; ++ks) {
    const v16h a = load_a_rm(sC, 512, 0, ks * 32, lane);
#pragma unroll
    for (int j = 0; j < 4; ++j) {
      const int nt = wave * 4 + j;
      acc[j] = wmma_f16f32(a, load_b_nm(g1T, 512, nt * 16, ks * 32, lane), acc[j]);
    }
  }
#pragma unroll
  for (int j = 0; j < 4; ++j) {
    const int nt = wave * 4 + j;
#pragma unroll
    for (int i = 0; i < 8; ++i) {
      const int r = kh * 8 + i, c = nt * 16 + ln;
      sH[r * 256 + c] = fmaxf(acc[j][i] + g1b[c], 0.f);
    }
  }
  __syncthreads();
  if (tid < 64) {
    const int r = tid >> 2, j = tid & 3;
    float s = g2b[j];
#pragma unroll 8
    for (int c = 0; c < 256; ++c) s += sH[r * 256 + c] * g2w[c * 4 + j];
    sG[r * 4 + j] = s;
  }
  __syncthreads();
  if (tid < 16) {
    const float a = sG[tid * 4 + 0], b = sG[tid * 4 + 1];
    const float c = sG[tid * 4 + 2], d = sG[tid * 4 + 3];
    const float m = fmaxf(fmaxf(a, b), fmaxf(c, d));
    const float ea = __expf(a - m), eb = __expf(b - m);
    const float ec = __expf(c - m), ed = __expf(d - m);
    const float inv = 1.0f / (ea + eb + ec + ed);
    sG[tid * 4 + 0] = ea * inv;
    sG[tid * 4 + 1] = eb * inv;
    sG[tid * 4 + 2] = ec * inv;
    sG[tid * 4 + 3] = ed * inv;
  }
  __syncthreads();
  for (int t = tid; t < 16 * DD; t += 128) {
    const int r = t >> 7, d = t & 127;
    const size_t idx = (size_t)(rowbase + r) * DD + d;
    float v;
    if (anyN[rowbase + r] > 0.f)
      v = sG[r * 4 + 0] * E[idx] + sG[r * 4 + 1] * posW[idx] +
          sG[r * 4 + 2] * bndW[idx] + sG[r * 4 + 3] * negW[idx];
    else
      v = E[idx];
    agg[idx] = v;
  }
}

// =====================================================================
// Kernel D: cur = relu(agg @ gc_w[l] + gc_b[l]); out (+)= softmax(lw)[l]*cur
// =====================================================================
__global__ __launch_bounds__(128) void gc_layer_kernel(
    const float* __restrict__ aggM, const float* __restrict__ gw,
    const float* __restrict__ gb, const float* __restrict__ layerW,
    int l, float* __restrict__ out, float* __restrict__ Ecur) {
  __shared__ float sA[DD];
  const int r = blockIdx.x;
  const int d = threadIdx.x;
  sA[d] = aggM[(size_t)r * DD + d];
  __syncthreads();
  float s = gb[d];
#pragma unroll 8
  for (int i = 0; i < DD; ++i) s += sA[i] * gw[i * DD + d];
  s = fmaxf(s, 0.f);
  Ecur[(size_t)r * DD + d] = s;
  const float w0 = layerW[0], w1 = layerW[1];
  const float m = fmaxf(w0, w1);
  const float e0 = __expf(w0 - m), e1 = __expf(w1 - m);
  const float lw = ((l == 0) ? e0 : e1) / (e0 + e1);
  const size_t idx = (size_t)r * DD + d;
  if (l == 0) out[idx] = lw * s;
  else out[idx] += lw * s;
}

// g1_w [512][256] f32 -> g1T f16 [256][512] (out-major so B-frags are contiguous)
__global__ __launch_bounds__(256) void prep_g1T_kernel(const float* __restrict__ g1w,
                                                       f16* __restrict__ g1T) {
  const int t = blockIdx.x * 256 + threadIdx.x;
  if (t < 512 * 256) {
    const int i = t >> 8, o = t & 255;
    g1T[(size_t)o * 512 + i] = (f16)g1w[t];
  }
}

extern "C" void kernel_launch(void* const* d_in, const int* in_sizes, int n_in,
                              void* d_out, int out_size, void* d_ws, size_t ws_size,
                              hipStream_t stream) {
  (void)in_sizes; (void)n_in; (void)out_size; (void)ws_size;
  const float* adj = (const float*)d_in[0];
  const float* E0 = (const float*)d_in[1];
  const float* gc_w = (const float*)d_in[2];
  const float* gc_b = (const float*)d_in[3];
  const float* wq = (const float*)d_in[4];
  const float* bq = (const float*)d_in[5];
  const float* wk = (const float*)d_in[6];
  const float* bk = (const float*)d_in[7];
  const float* wv = (const float*)d_in[8];
  const float* bv = (const float*)d_in[9];
  const float* g1w = (const float*)d_in[10];
  const float* g1b = (const float*)d_in[11];
  const float* g2w = (const float*)d_in[12];
  const float* g2b = (const float*)d_in[13];
  const float* lw = (const float*)d_in[14];
  float* out = (float*)d_out;

  char* ws = (char*)d_ws;
  size_t off = 0;
  auto take = [&](size_t bytes) -> void* {
    void* p = ws + off;
    off += (bytes + 255) & ~(size_t)255;
    return p;
  };
  float* Ecur = (float*)take((size_t)NN * DD * 4);
  f16* Eh = (f16*)take((size_t)NN * DD * 2);
  f16* EhT = (f16*)take((size_t)NN * DD * 2);
  f16* Qh = (f16*)take((size_t)NN * DD * 2);
  f16* Kh = (f16*)take((size_t)NN * DD * 2);
  f16* VhT = (f16*)take((size_t)NN * DD * 2);
  float* invn = (float*)take((size_t)NN * 4);
  float* anyN = (float*)take((size_t)NN * 4);
  float* posW = (float*)take((size_t)NN * DD * 4);
  float* bndW = (float*)take((size_t)NN * DD * 4);
  float* negW = (float*)take((size_t)NN * DD * 4);
  float* agg = (float*)take((size_t)NN * DD * 4);
  f16* g1T = (f16*)take((size_t)512 * 256 * 2);

  prep_g1T_kernel<<<512, 256, 0, stream>>>(g1w, g1T);

  const float* Ein = E0;
  for (int l = 0; l < 2; ++l) {
    qkv_norm_kernel<<<NN, 128, 0, stream>>>(Ein, wq, bq, wk, bk, wv, bv,
                                            Eh, EhT, Qh, Kh, VhT, invn);
    fused_aggregate_kernel<<<NN / 16, 64, 0, stream>>>(adj, Eh, EhT, Qh, Kh, VhT,
                                                       invn, posW, bndW, negW, anyN);
    gate_fuse_kernel<<<NN / 16, 128, 0, stream>>>(Ein, posW, bndW, negW, anyN,
                                                  g1T, g1b, g2w, g2b, agg);
    gc_layer_kernel<<<NN, 128, 0, stream>>>(agg, gc_w + (size_t)l * DD * DD,
                                            gc_b + (size_t)l * DD, lw, l, out, Ecur);
    Ein = Ecur;
  }
}
